// PiecewiseRationalQuadraticCouplingTransform_55963423867165
// MI455X (gfx1250) — matrix operand order repaired and checked by reading, not compiled, not run on hardware
//
#include <hip/hip_runtime.h>

typedef __attribute__((ext_vector_type(16))) __bf16 v16bf;
typedef __attribute__((ext_vector_type(8)))  float  v8f;
typedef __attribute__((ext_vector_type(4)))  int    i4;

#define NUM_BINS   10
#define HIDDEN     512
#define FEATS      64
#define PPF        29            // params per transform feature
#define PARAM_DIM  928
#define NT         58            // N tiles of 16 covering 928
#define NTP        60            // padded tiles (4 waves x 15)
#define KSTEPS     16            // 512 / 32
#define H_STRIDE   520           // bf16 per h row (512 + 8 pad -> bank friendly)
#define P_STRIDE   932           // f32 per param row (928 + 4 pad)
#define SMEM_BYTES (16 * P_STRIDE * 4)   // 59648 < 64KB; h region (33280B) aliases
#define W1F_ELEMS  (32 * 32 * 16)
#define W2F_ELEMS  (KSTEPS * NTP * 32 * 16)

union frag_u { v16bf v; i4 q[2]; };

// ---------------------------------------------------------------------------
// Prep: convert W1/W2 (f32) to bf16 in WMMA B-fragment order.
// B-matrix (KxN) fragment layout assumed (mirrors documented A layout / sparse
// B tables): lane l holds column N = tile*16 + (l&15); element e holds
// K = (e&7) + 8*(l>>4) + 16*(e>>3) within the 32-wide K step.
// ---------------------------------------------------------------------------
__global__ __launch_bounds__(256) void nsf_prep(const float* __restrict__ W1,
                                                const float* __restrict__ W2,
                                                unsigned short* __restrict__ w1f_,
                                                unsigned short* __restrict__ w2f_) {
  __bf16* w1f = (__bf16*)w1f_;
  __bf16* w2f = (__bf16*)w2f_;
  int idx = blockIdx.x * 256 + threadIdx.x;
  if (idx < W1F_ELEMS) {
    int tile = idx >> 9, rem = idx & 511;
    int lane = rem >> 4, e = rem & 15;
    int K  = (e & 7) + ((lane >> 4) << 3) + ((e >> 3) << 4);
    int nn = (tile << 4) + (lane & 15);
    w1f[idx] = (__bf16)W1[K * HIDDEN + nn];
  } else if (idx < W1F_ELEMS + W2F_ELEMS) {
    int t = idx - W1F_ELEMS;
    int kk  = t / (NTP * 512);
    int rem = t - kk * (NTP * 512);
    int tile = rem >> 9; int rem2 = rem & 511;
    int lane = rem2 >> 4, e = rem2 & 15;
    int K  = kk * 32 + (e & 7) + ((lane >> 4) << 3) + ((e >> 3) << 4);
    int nn = (tile << 4) + (lane & 15);
    float v = (nn < PARAM_DIM) ? W2[K * PARAM_DIM + nn] : 0.0f;
    w2f[t] = (__bf16)v;
  }
}

__device__ __forceinline__ float softplusf(float v) {
  return (v > 15.0f) ? v : log1pf(__expf(v));
}

// ---------------------------------------------------------------------------
// Fused: GEMM1 (relu MLP layer) -> LDS -> GEMM2 (param head) -> spline.
// 256 threads = 8 waves, 32 rows per block.
// ---------------------------------------------------------------------------
__global__ __launch_bounds__(256) void nsf_fused(const float* __restrict__ x,
        const float* __restrict__ b1, const float* __restrict__ b2,
        const unsigned short* __restrict__ w1f_, const unsigned short* __restrict__ w2f_,
        float* __restrict__ out, float* __restrict__ lad_out,
        float* __restrict__ cost_out) {
  __shared__ __align__(16) char smem[SMEM_BYTES];
  const __bf16* w1f = (const __bf16*)w1f_;
  const __bf16* w2f = (const __bf16*)w2f_;

  const int lane   = threadIdx.x & 31;
  const int wv     = threadIdx.x >> 5;
  const int mtile  = wv >> 2;        // which 16-row half of the 32-row block
  const int q      = wv & 3;         // quarter of the N-tiles
  const int lhalf  = lane >> 4;
  const int l16    = lane & 15;
  const int baseRow = blockIdx.x * 32;

  // ---------------- GEMM1: h = relu(ident @ W1 + b1), h -> LDS (bf16) -------
  frag_u a1;
  {
    const float* xr = x + (size_t)(baseRow + mtile * 16 + l16) * FEATS;
#pragma unroll
    for (int e = 0; e < 16; ++e) {
      int f = ((e < 8) ? e : (e + 8)) + (lhalf << 3);   // ident feature index
      a1.v[e] = (__bf16)xr[2 * f + 1];                  // identity cols are odd
    }
  }
  __bf16* hsm = (__bf16*)smem;
#pragma unroll
  for (int i = 0; i < 8; ++i) {
    int n = (q << 3) + i + ((mtile) ? 0 : 0);           // 32 tiles over 4 waves, x2 redundancy by mtile
    // waves 0-3 and 4-7 each cover n = q*8+i for their own mtile's rows
    frag_u bf;
    const i4* bp = (const i4*)(w1f + ((n * 32 + lane) << 4));
    bf.q[0] = bp[0]; bf.q[1] = bp[1];
    v8f c = {};
    c = __builtin_amdgcn_wmma_f32_16x16x32_bf16(false, a1.v, false, bf.v,
                                                (short)0, c, false, false);
    int col = (n << 4) + l16;
    float bias = b1[col];
#pragma unroll
    for (int r = 0; r < 8; ++r) {
      int rowl = mtile * 16 + r + (lhalf << 3);
      float hv = fmaxf(c[r] + bias, 0.0f);
      hsm[rowl * H_STRIDE + col] = (__bf16)hv;
    }
  }
  __syncthreads();

  // ---------------- GEMM2: params = h @ W2 (+b2 later) ----------------------
  v8f acc[15];
#pragma unroll
  for (int j = 0; j < 15; ++j) acc[j] = {};
  {
    const int rowl = mtile * 16 + l16;
    const char* hrow = smem + rowl * (H_STRIDE * 2);
#pragma unroll 1
    for (int kk = 0; kk < KSTEPS; ++kk) {
      frag_u a;
      const i4* ap = (const i4*)(hrow + kk * 64 + (lhalf << 4));
      a.q[0] = ap[0];       // K = kk*32 + 8*lhalf + [0..7]
      a.q[1] = ap[2];       // K = kk*32 + 16 + 8*lhalf + [0..7]
#pragma unroll
      for (int j = 0; j < 15; ++j) {
        int n = q * 15 + j;
        frag_u bf;
        const i4* bp = (const i4*)(w2f + (((kk * NTP + n) * 32 + lane) << 4));
        bf.q[0] = bp[0]; bf.q[1] = bp[1];
        acc[j] = __builtin_amdgcn_wmma_f32_16x16x32_bf16(false, a.v, false, bf.v,
                                                         (short)0, acc[j], false, false);
      }
    }
  }
  __syncthreads();   // all h reads done; param region may now alias h region

  float* pr = (float*)smem;
  const float SCALE = 0.044194173824159216f;  // 1/sqrt(512)

#pragma unroll 1
  for (int phase = 0; phase < 2; ++phase) {
    // ---- stage this 16-row tile's params (f32, +b2) into LDS ----
    if (mtile == phase) {
#pragma unroll
      for (int j = 0; j < 15; ++j) {
        int n = q * 15 + j;
        if (n < NT) {
          int p = (n << 4) + l16;     // p <= 927
          float bias = b2[p];
#pragma unroll
          for (int r = 0; r < 8; ++r) {
            int rowt = r + (lhalf << 3);
            pr[rowt * P_STRIDE + p] = acc[j][r] + bias;
          }
        }
      }
    }
    __syncthreads();

    // ---- spline: wave handles 2 rows, lane = transform feature ----
#pragma unroll 1
    for (int rr = 0; rr < 2; ++rr) {
      int rowt = wv * 2 + rr;
      int rowg = baseRow + phase * 16 + rowt;
      const float* xr = x + (size_t)rowg * FEATS;
      float* orow = out + (size_t)rowg * FEATS;
      int f = lane;
      float xv  = xr[2 * f];
      float xid = xr[2 * f + 1];
      orow[2 * f + 1] = xid;

      const float* pp = pr + rowt * P_STRIDE + f * PPF;
      float uw[NUM_BINS], uh[NUM_BINS], ud[NUM_BINS - 1];
#pragma unroll
      for (int i = 0; i < NUM_BINS; ++i) uw[i] = pp[i] * SCALE;
#pragma unroll
      for (int i = 0; i < NUM_BINS; ++i) uh[i] = pp[NUM_BINS + i] * SCALE;
#pragma unroll
      for (int i = 0; i < NUM_BINS - 1; ++i) ud[i] = pp[2 * NUM_BINS + i];

      float cumw[NUM_BINS + 1], cumh[NUM_BINS + 1];
      {
        float mx = uw[0];
#pragma unroll
        for (int i = 1; i < NUM_BINS; ++i) mx = fmaxf(mx, uw[i]);
        float ex[NUM_BINS]; float s = 0.0f;
#pragma unroll
        for (int i = 0; i < NUM_BINS; ++i) { ex[i] = __expf(uw[i] - mx); s += ex[i]; }
        float inv = 1.0f / s, cs = 0.0f;
        cumw[0] = -1.0f;
#pragma unroll
        for (int i = 0; i < NUM_BINS; ++i) {
          cs += 0.001f + (1.0f - 0.001f * NUM_BINS) * ex[i] * inv;
          cumw[i + 1] = 2.0f * cs - 1.0f;
        }
        cumw[NUM_BINS] = 1.0f;
      }
      {
        float mx = uh[0];
#pragma unroll
        for (int i = 1; i < NUM_BINS; ++i) mx = fmaxf(mx, uh[i]);
        float ex[NUM_BINS]; float s = 0.0f;
#pragma unroll
        for (int i = 0; i < NUM_BINS; ++i) { ex[i] = __expf(uh[i] - mx); s += ex[i]; }
        float inv = 1.0f / s, cs = 0.0f;
        cumh[0] = -1.0f;
#pragma unroll
        for (int i = 0; i < NUM_BINS; ++i) {
          cs += 0.001f + (1.0f - 0.001f * NUM_BINS) * ex[i] * inv;
          cumh[i + 1] = 2.0f * cs - 1.0f;
        }
        cumh[NUM_BINS] = 1.0f;
      }
      float d[NUM_BINS + 1];
      d[0] = 1.0f; d[NUM_BINS] = 1.0f;   // MIN_D + softplus(const) == 1 exactly
#pragma unroll
      for (int i = 0; i < NUM_BINS - 1; ++i) d[i + 1] = 0.001f + softplusf(ud[i]);

      bool inside = (xv >= -1.0f) && (xv <= 1.0f);
      float xc = fminf(fmaxf(xv, -1.0f), 1.0f);
      int bin = -1;
#pragma unroll
      for (int k = 0; k <= NUM_BINS; ++k) {
        float loc = (k == NUM_BINS) ? (1.0f + 1e-6f) : cumw[k];
        bin += (xc >= loc) ? 1 : 0;
      }
      bin = max(0, min(NUM_BINS - 1, bin));
      float in_cw = 0.f, in_w = 1.f, in_ch = 0.f, in_h = 1.f, dk = 1.f, dk1 = 1.f;
#pragma unroll
      for (int k = 0; k < NUM_BINS; ++k) {
        bool m = (k == bin);
        in_cw = m ? cumw[k] : in_cw;
        in_w  = m ? (cumw[k + 1] - cumw[k]) : in_w;
        in_ch = m ? cumh[k] : in_ch;
        in_h  = m ? (cumh[k + 1] - cumh[k]) : in_h;
        dk    = m ? d[k] : dk;
        dk1   = m ? d[k + 1] : dk1;
      }
      float in_delta = in_h / in_w;
      float theta = (xc - in_cw) / in_w;
      float t1m = theta * (1.0f - theta);
      float numer = in_h * (in_delta * theta * theta + dk * t1m);
      float denom = in_delta + (dk + dk1 - 2.0f * in_delta) * t1m;
      float y = in_ch + numer / denom;
      float omt = 1.0f - theta;
      float dnum = in_delta * in_delta *
                   (dk1 * theta * theta + 2.0f * in_delta * t1m + dk * omt * omt);
      float lad = __logf(dnum) - 2.0f * __logf(denom);

      float yout = inside ? y : xv;
      float ladv = inside ? lad : 0.0f;
      orow[2 * f] = yout;
      float df = xv - yout;
      float costv = df * df;
#pragma unroll
      for (int off = 16; off > 0; off >>= 1) {
        ladv  += __shfl_xor(ladv, off);
        costv += __shfl_xor(costv, off);
      }
      if (lane == 0) { lad_out[rowg] = ladv; cost_out[rowg] = costv; }
    }
    __syncthreads();
  }
}

extern "C" void kernel_launch(void* const* d_in, const int* in_sizes, int n_in,
                              void* d_out, int out_size, void* d_ws, size_t ws_size,
                              hipStream_t stream) {
  const float* x  = (const float*)d_in[0];
  const float* W1 = (const float*)d_in[1];
  const float* b1 = (const float*)d_in[2];
  const float* W2 = (const float*)d_in[3];
  const float* b2 = (const float*)d_in[4];
  int Bn = in_sizes[0] / FEATS;

  unsigned short* w1f = (unsigned short*)d_ws;
  unsigned short* w2f = w1f + W1F_ELEMS;

  float* out  = (float*)d_out;
  float* lad  = out + (size_t)Bn * FEATS;
  float* cost = lad + Bn;

  int prep_blocks = (W1F_ELEMS + W2F_ELEMS + 255) / 256;
  nsf_prep<<<prep_blocks, 256, 0, stream>>>(W1, W2, w1f, w2f);
  nsf_fused<<<Bn / 32, 256, 0, stream>>>(x, b1, b2, w1f, w2f, out, lad, cost);
}
